// MultiHeadGraphAttention_74148315398564
// MI455X (gfx1250) — compile-verified, matrix operand order checked by reading, not run on hardware
//
#include <hip/hip_runtime.h>

// MultiHeadGraphAttention for MI455X (gfx1250, wave32, WMMA f16->f32).
// B=8, N=1024, F_IN=F_OUT=128, H=8, D_K=16.

typedef __attribute__((ext_vector_type(16))) _Float16 v16h;
typedef __attribute__((ext_vector_type(8)))  float    v8f;

#define BB   8
#define NN   1024
#define FIN  128
#define FOUT 128
#define HH   8
#define DK   16

union F16x8 { uint4 u4; _Float16 h[8]; };

__device__ __forceinline__ v8f wmma_f16(v16h a, v16h b, v8f c) {
  // D = A(16x32 f16) x B(32x16 f16) + C(16x16 f32)
  return __builtin_amdgcn_wmma_f32_16x16x32_f16(
      /*neg_a=*/false, a, /*neg_b=*/false, b,
      /*c_mod=*/(short)0, c, /*reuse_a=*/false, /*reuse_b=*/false);
}

// ---------------------------------------------------------------------------
// Kernel 1: fused QKV projection. One wave -> one 16x16 output tile.
//   sel 0: Q -> Qh[B,H,N,16] f16
//   sel 1: K -> Kh[B,H,N,16] f16
//   sel 2: V -> Vt[B,H,16,N] f16  (transposed for contiguous B-frag loads)
// ---------------------------------------------------------------------------
__global__ __launch_bounds__(32)
void gat_proj_qkv(const float* __restrict__ X,
                  const float* __restrict__ Wq, const float* __restrict__ bq,
                  const float* __restrict__ Wk, const float* __restrict__ bk,
                  const float* __restrict__ Wv, const float* __restrict__ bv,
                  _Float16* __restrict__ Qh, _Float16* __restrict__ Kh,
                  _Float16* __restrict__ Vt)
{
  const int lane = threadIdx.x;
  const int r    = lane & 15;
  const int half = lane >> 4;
  const int t    = blockIdx.x;   // 16-row tile over B*N (0..511)
  const int h    = blockIdx.y;   // head == 16-wide output chunk
  const int sel  = blockIdx.z;   // 0=Q 1=K 2=V

  const float* W  = (sel == 0) ? Wq : (sel == 1) ? Wk : Wv;
  const float* bi = (sel == 0) ? bq : (sel == 1) ? bk : bv;

  const int arow = t * 16 + r;        // A-frag row (node)
  const int ocol = h * DK + r;        // B-frag column (output channel)

  v8f acc = {0.f, 0.f, 0.f, 0.f, 0.f, 0.f, 0.f, 0.f};

#pragma unroll
  for (int kk = 0; kk < 4; ++kk) {    // K = 128 in chunks of 32
    const float* xa = X + (size_t)arow * FIN + kk * 32 + half * 8;
    const float* wb = W + (size_t)ocol * FIN + kk * 32 + half * 8;
    v16h a, bfr;
#pragma unroll
    for (int e = 0; e < 8; ++e) {
      a[e]       = (_Float16)xa[e];        // K = kk*32 + half*8 + e
      a[8 + e]   = (_Float16)xa[16 + e];   // K = kk*32 + 16 + half*8 + e
      bfr[e]     = (_Float16)wb[e];
      bfr[8 + e] = (_Float16)wb[16 + e];
    }
    acc = wmma_f16(a, bfr, acc);
  }

  const float bias = bi[ocol];

#pragma unroll
  for (int v = 0; v < 8; ++v) {
    const int m    = v + half * 8;        // C row
    const int node = t * 16 + m;          // tile never crosses batch (N%16==0)
    const int b    = node >> 10;
    const int n    = node & (NN - 1);
    const float val = acc[v] + bias;
    if (sel == 0)
      Qh[((size_t)(b * HH + h) * NN + n) * DK + r] = (_Float16)val;
    else if (sel == 1)
      Kh[((size_t)(b * HH + h) * NN + n) * DK + r] = (_Float16)val;
    else
      Vt[((size_t)(b * HH + h) * DK + r) * NN + n] = (_Float16)val;
  }
}

// ---------------------------------------------------------------------------
// Kernel 2: masked attention, flash-style streaming softmax.
// Block = 256 threads = 8 waves = 8 heads over one (batch, 16-query tile),
// so adjacency reads are shared across heads through WGP$/L2.
// ---------------------------------------------------------------------------
__global__ __launch_bounds__(256)
void gat_attn(const _Float16* __restrict__ Qh, const _Float16* __restrict__ Kh,
              const _Float16* __restrict__ Vt, const float* __restrict__ Adj,
              _Float16* __restrict__ AO)
{
  __shared__ __align__(16) _Float16 pb[HH * 16 * 32];  // per-wave P buffers

  const int lane = threadIdx.x & 31;
  const int h    = threadIdx.x >> 5;   // wave id == head
  const int r    = lane & 15;
  const int half = lane >> 4;
  const int qt   = blockIdx.x;         // query tile within batch (0..63)
  const int b    = blockIdx.y;
  const int bh   = b * HH + h;

  _Float16* P = pb + h * (16 * 32);

  // Q A-fragment (16x32, K-dim d padded 16 -> 32 with zeros)
  v16h qa;
  {
    F16x8 t0;
    t0.u4 = *reinterpret_cast<const uint4*>(
        Qh + ((size_t)bh * NN + qt * 16 + r) * DK + half * 8);
#pragma unroll
    for (int e = 0; e < 8; ++e) { qa[e] = t0.h[e]; qa[8 + e] = (_Float16)0.f; }
  }

  v8f oacc = {0.f, 0.f, 0.f, 0.f, 0.f, 0.f, 0.f, 0.f};
  float mrun[8], lrun[8];
#pragma unroll
  for (int v = 0; v < 8; ++v) { mrun[v] = -1.0e30f; lrun[v] = 0.f; }

  const float* arowbase = Adj + (size_t)b * NN * NN + (size_t)(qt * 16) * NN;

  for (int kt = 0; kt < NN / 32; ++kt) {
    const int kbase = kt * 32;

    // K^T B-fragments for keys [kbase, kbase+16) and [kbase+16, kbase+32)
    v16h kb0, kb1;
    {
      const _Float16* kp = Kh + ((size_t)bh * NN + kbase + r) * DK + half * 8;
      F16x8 t0; t0.u4 = *reinterpret_cast<const uint4*>(kp);
      F16x8 t1; t1.u4 = *reinterpret_cast<const uint4*>(kp + 16 * DK);
#pragma unroll
      for (int e = 0; e < 8; ++e) {
        kb0[e] = t0.h[e]; kb0[8 + e] = (_Float16)0.f;
        kb1[e] = t1.h[e]; kb1[8 + e] = (_Float16)0.f;
      }
    }

    const v8f zero = {0.f, 0.f, 0.f, 0.f, 0.f, 0.f, 0.f, 0.f};
    v8f s0 = wmma_f16(qa, kb0, zero);   // S[:, 0:16]
    v8f s1 = wmma_f16(qa, kb1, zero);   // S[:, 16:32]

    // Mask, scale (1/sqrt(16)=0.25), streaming softmax update.
#pragma unroll
    for (int v = 0; v < 8; ++v) {
      const int m = v + half * 8;                         // C-frag row
      const float a0 = arowbase[(size_t)m * NN + kbase + r];
      const float a1 = arowbase[(size_t)m * NN + kbase + 16 + r];
      const float sv0 = (a0 > 0.f) ? s0[v] * 0.25f : -1.0e9f;
      const float sv1 = (a1 > 0.f) ? s1[v] * 0.25f : -1.0e9f;

      float tm = fmaxf(sv0, sv1);
#pragma unroll
      for (int off = 8; off >= 1; off >>= 1)
        tm = fmaxf(tm, __shfl_xor(tm, off, 32));          // 16-lane row max

      const float mn = fmaxf(mrun[v], tm);
      const float al = __expf(mrun[v] - mn);
      const float e0 = __expf(sv0 - mn);
      const float e1 = __expf(sv1 - mn);

      float rs = e0 + e1;
#pragma unroll
      for (int off = 8; off >= 1; off >>= 1)
        rs += __shfl_xor(rs, off, 32);                    // 16-lane row sum

      lrun[v] = lrun[v] * al + rs;
      mrun[v] = mn;
      oacc[v] *= al;

      P[m * 32 + r]      = (_Float16)e0;                  // C-layout -> LDS
      P[m * 32 + 16 + r] = (_Float16)e1;
    }

    // DS ops from one wave execute in order; wait for LDS writes + fence
    // the compiler so the A-layout reads below stay behind the stores.
    asm volatile("s_wait_dscnt 0" ::: "memory");

    // P as A-fragment (16x32) from LDS, V as B-fragment (32x16) from Vt.
    v16h pa, vb;
    {
      const _Float16* pp = P + r * 32 + half * 8;
      F16x8 t0; t0.u4 = *reinterpret_cast<const uint4*>(pp);
      F16x8 t1; t1.u4 = *reinterpret_cast<const uint4*>(pp + 16);
      const _Float16* vp = Vt + ((size_t)bh * DK + r) * NN + kbase + half * 8;
      F16x8 t2; t2.u4 = *reinterpret_cast<const uint4*>(vp);
      F16x8 t3; t3.u4 = *reinterpret_cast<const uint4*>(vp + 16);
#pragma unroll
      for (int e = 0; e < 8; ++e) {
        pa[e] = t0.h[e]; pa[8 + e] = t1.h[e];
        vb[e] = t2.h[e]; vb[8 + e] = t3.h[e];
      }
    }
    oacc = wmma_f16(pa, vb, oacc);      // O += P(16x32) x V(32x16)
  }

  // Normalize and write attention output [B,N,H*DK] f16.
#pragma unroll
  for (int v = 0; v < 8; ++v) {
    const int m = v + half * 8;
    const float val = oacc[v] / lrun[v];   // diagonal edge => lrun > 0
    AO[((size_t)b * NN + qt * 16 + m) * FOUT + h * DK + r] = (_Float16)val;
  }
}

// ---------------------------------------------------------------------------
// Kernel 3: output projection Y = AO @ Wo^T + bo (f32 result).
// ---------------------------------------------------------------------------
__global__ __launch_bounds__(32)
void gat_outproj(const _Float16* __restrict__ AO, const float* __restrict__ Wo,
                 const float* __restrict__ bo, float* __restrict__ Y)
{
  const int lane = threadIdx.x;
  const int r    = lane & 15;
  const int half = lane >> 4;
  const int t    = blockIdx.x;   // 16-row tile over B*N
  const int oc   = blockIdx.y;   // 16-wide output chunk

  const int arow = t * 16 + r;
  const int ocol = oc * DK + r;

  v8f acc = {0.f, 0.f, 0.f, 0.f, 0.f, 0.f, 0.f, 0.f};

#pragma unroll
  for (int kk = 0; kk < 4; ++kk) {
    const _Float16* ap = AO + (size_t)arow * FOUT + kk * 32 + half * 8;
    const float*    wb = Wo + (size_t)ocol * FOUT + kk * 32 + half * 8;
    F16x8 t0; t0.u4 = *reinterpret_cast<const uint4*>(ap);
    F16x8 t1; t1.u4 = *reinterpret_cast<const uint4*>(ap + 16);
    v16h a, bfr;
#pragma unroll
    for (int e = 0; e < 8; ++e) {
      a[e]       = t0.h[e];
      a[8 + e]   = t1.h[e];
      bfr[e]     = (_Float16)wb[e];
      bfr[8 + e] = (_Float16)wb[16 + e];
    }
    acc = wmma_f16(a, bfr, acc);
  }

  const float bias = bo[ocol];
#pragma unroll
  for (int v = 0; v < 8; ++v) {
    const int m = v + half * 8;
    Y[(size_t)(t * 16 + m) * FOUT + oc * DK + r] = acc[v] + bias;
  }
}

// ---------------------------------------------------------------------------
extern "C" void kernel_launch(void* const* d_in, const int* in_sizes, int n_in,
                              void* d_out, int out_size, void* d_ws, size_t ws_size,
                              hipStream_t stream) {
  const float* X   = (const float*)d_in[0];
  const float* Adj = (const float*)d_in[1];
  const float* Wq  = (const float*)d_in[2];
  const float* bq  = (const float*)d_in[3];
  const float* Wk  = (const float*)d_in[4];
  const float* bk  = (const float*)d_in[5];
  const float* Wv  = (const float*)d_in[6];
  const float* bv  = (const float*)d_in[7];
  const float* Wo  = (const float*)d_in[8];
  const float* bo  = (const float*)d_in[9];
  float* Y = (float*)d_out;

  // Workspace layout (8 MiB total, fully overwritten each call):
  //   [0,2M)   Qh  f16 [B,H,N,16]
  //   [2M,4M)  Kh  f16 [B,H,N,16]
  //   [4M,6M)  Vt  f16 [B,H,16,N]
  //   [6M,8M)  AO  f16 [B,N,128]
  char* ws = (char*)d_ws;
  _Float16* Qh = (_Float16*)(ws);
  _Float16* Kh = (_Float16*)(ws + (2u << 20));
  _Float16* Vt = (_Float16*)(ws + (4u << 20));
  _Float16* AO = (_Float16*)(ws + (6u << 20));

  // 1) QKV projections: (B*N/16)=512 row tiles x 8 heads x {Q,K,V}
  gat_proj_qkv<<<dim3(512, 8, 3), 32, 0, stream>>>(
      X, Wq, bq, Wk, bk, Wv, bv, Qh, Kh, Vt);

  // 2) Masked flash attention: (N/16)=64 query tiles x B, 8 waves = 8 heads
  gat_attn<<<dim3(64, 8), 256, 0, stream>>>(Qh, Kh, Vt, Adj, AO);

  // 3) Output projection
  gat_outproj<<<dim3(512, 8), 32, 0, stream>>>(AO, Wo, bo, Y);
}
